// BioacousticHORN_12025908429346
// MI455X (gfx1250) — matrix operand order repaired and checked by reading, not compiled, not run on hardware
//
#include <hip/hip_runtime.h>
#include <hip/hip_bf16.h>

// ---------------------------------------------------------------------------
// BioacousticHORN on MI455X (gfx1250, wave32, WMMA + async-to-LDS)
//
// Layers are feed-forward (no cross-layer feedback), so the T=512 "scan"
// decomposes into elementwise recurrences + two large GEMMs:
//   K0: convert+transpose weights to f16 [N][K] (one-time, ~1MB)
//   K1: input-oscillator scan  -> x_in_all (fp32, d_out) + x_in_h (f16, ws)
//   K2: WMMA GEMM [32768,128]@[128,2048] + bias -> proj (f16, ws)
//   K3: fused reservoir+output scan -> x_res_all, x_out_all (d_out) + x_out_h
//   K4: WMMA GEMM [32768,2048]@[2048,128] + bias -> preds (fp32, d_out)
//
// GEMM: 256-thread blocks (8 waves), 64(M)x128(N) block tile, each wave a
// 32x32 tile (2x2 WMMA accumulators -> 2:1 ds_load:wmma with fragment reuse).
// Staging via GLOBAL_LOAD_ASYNC_TO_LDS_B128 (ASYNCcnt), LDS double-buffered,
// exactly 6 async ops per thread per K-step so s_wait_asynccnt 6 == "previous
// tile complete" (async loads complete in order).
//
// Workspace requirement: ~266 MB (see kernel_launch layout).
// ---------------------------------------------------------------------------

typedef __attribute__((ext_vector_type(16))) _Float16 v16h;
typedef __attribute__((ext_vector_type(8)))  _Float16 v8h;
typedef __attribute__((ext_vector_type(8)))  float    v8f;

#define H_STEP  0.5f
#define CLAMP_V 5.0f

constexpr int B_ = 64, T_ = 512, D_ = 128, R_ = 2048;
constexpr int M_ = B_ * T_;   // 32768 rows for both GEMMs

__device__ __forceinline__ float clamp5(float v) {
    return fminf(fmaxf(v, -CLAMP_V), CLAMP_V);
}

// LDS byte address of a generic (__shared__) pointer: ISA 10.2 — LDS aperture
// addresses use addr[31:0] directly as the LDS offset.
__device__ __forceinline__ unsigned lds_addr(const void* p) {
    return (unsigned)(unsigned long long)p;
}

__device__ __forceinline__ void async_load_b128(unsigned lds_dst, const void* gsrc) {
    asm volatile("global_load_async_to_lds_b128 %0, %1, off"
                 :: "v"(lds_dst), "v"(gsrc) : "memory");
}

__device__ __forceinline__ v16h cat8(v8h lo, v8h hi) {
    return __builtin_shufflevector(lo, hi, 0,1,2,3,4,5,6,7,8,9,10,11,12,13,14,15);
}

// ---------------------------------------------------------------------------
// K0: fp32 weight [K][N] row-major -> f16 transposed [N][K]
// ---------------------------------------------------------------------------
__global__ __launch_bounds__(256)
void convert_transpose_f16(const float* __restrict__ src,
                           _Float16* __restrict__ dst, int K, int N) {
    int idx = blockIdx.x * 256 + threadIdx.x;
    if (idx >= K * N) return;
    int k = idx / N, n = idx % N;
    dst[(size_t)n * K + k] = (_Float16)src[idx];
}

// ---------------------------------------------------------------------------
// K1: input oscillator scan. One thread per (b,d); block = one batch row.
// ---------------------------------------------------------------------------
__global__ __launch_bounds__(128)
void scan_in_kernel(const float* __restrict__ inputs,
                    const float* __restrict__ om_p,
                    const float* __restrict__ ga_p,
                    const float* __restrict__ al_p,
                    float* __restrict__ x_in_all,
                    _Float16* __restrict__ x_in_h) {
    const int d = threadIdx.x;
    const int b = blockIdx.x;
    const float om = fabsf(om_p[d]);
    const float w2 = om * om;
    const float tg = 2.0f * fabsf(ga_p[d]);
    const float al = al_p[d];
    float x = 0.0f, y = 0.0f;
    const float* ip = inputs   + (size_t)b * T_ * D_ + d;
    float*       op = x_in_all + (size_t)b * T_ * D_ + d;
    _Float16*    hp = x_in_h   + (size_t)b * T_ * D_ + d;
    for (int t = 0; t < T_; ++t) {
        float f   = ip[(size_t)t * D_];
        float acc = al * f - tg * y - w2 * x;
        x = clamp5(x + H_STEP * y);
        y = clamp5(y + H_STEP * acc);
        op[(size_t)t * D_] = x;
        hp[(size_t)t * D_] = (_Float16)x;
    }
}

// ---------------------------------------------------------------------------
// K3: fused reservoir + output oscillator scan. One thread per (b,r).
// ---------------------------------------------------------------------------
__global__ __launch_bounds__(256)
void scan_res_out_kernel(const _Float16* __restrict__ proj_h,
                         const float* __restrict__ ro, const float* __restrict__ rg,
                         const float* __restrict__ ra,
                         const float* __restrict__ oo, const float* __restrict__ og,
                         const float* __restrict__ oa,
                         float* __restrict__ x_res_all,
                         float* __restrict__ x_out_all,
                         _Float16* __restrict__ x_out_h) {
    const int r = blockIdx.x * 256 + threadIdx.x;
    const int b = blockIdx.y;
    const float wr = fabsf(ro[r]); const float w2r = wr * wr;
    const float tgr = 2.0f * fabsf(rg[r]); const float ar = ra[r];
    const float wo = fabsf(oo[r]); const float w2o = wo * wo;
    const float tgo = 2.0f * fabsf(og[r]); const float ao = oa[r];
    float xr = 0.f, yr = 0.f, xo = 0.f, yo = 0.f;
    const size_t base = (size_t)b * T_ * R_ + r;
    for (int t = 0; t < T_; ++t) {
        const size_t idx = base + (size_t)t * R_;
        float f    = (float)proj_h[idx];
        float acc  = ar * f - tgr * yr - w2r * xr;
        xr = clamp5(xr + H_STEP * yr);
        yr = clamp5(yr + H_STEP * acc);
        x_res_all[idx] = xr;
        float acc2 = ao * xr - tgo * yo - w2o * xo;   // force = NEW x_res
        xo = clamp5(xo + H_STEP * yo);
        yo = clamp5(yo + H_STEP * acc2);
        x_out_all[idx] = xo;
        x_out_h[idx]   = (_Float16)xo;
    }
}

// ---------------------------------------------------------------------------
// K2/K4: WMMA GEMM.  C[M,N] = Ah[M,K](f16) * Bt[N,K](f16)^T + bias.
// Fragment layouts per CDNA5 ISA §7.12.2:
//   A 16x32 f16:  lane<16 row=lane K{0..7,16..23}; lane>=16 K{8..15,24..31}
//   B 32x16 f16:  lane<16 col=lane K{0..7,16..23}; lane>=16 K{8..15,24..31}
//   C/D 16x16 f32: vgpr i -> M=i (lanes 0-15, N=lane) / M=i+8 (lanes 16-31)
// ---------------------------------------------------------------------------
template <bool OUT_HALF>
__global__ __launch_bounds__(256)
void wmma_gemm_kernel(const _Float16* __restrict__ Ah,  // [M x K] f16
                      const _Float16* __restrict__ Bt,  // [N x K] f16
                      const float* __restrict__ bias,   // [N] fp32
                      void* __restrict__ Cout,          // [M x N]
                      int K, int N) {
    // rows padded to 72 halfs (144B) to de-stride banks; 16B-aligned reads.
    __shared__ __align__(16) _Float16 Ash[2][64 * 72];    // [m][k]
    __shared__ __align__(16) _Float16 Bsh[2][128 * 72];   // [n][k]

    const int tid  = threadIdx.x;
    const int lane = tid & 31;
    const int wave = tid >> 5;
    const int wm = wave >> 2;            // 0..1 : 32-row slice
    const int wn = wave & 3;             // 0..3 : 32-col slice
    const int m0 = blockIdx.y * 64;
    const int n0 = blockIdx.x * 128;

    // cooperative async-staging indices (uniform: 2 A + 4 B b128 per thread)
    const int a_row = tid >> 3, a_c = tid & 7;   // rows a_row, a_row+32
    const int b_row = tid >> 2, b_c = tid & 3;   // rows b_row, b_row+64; chunks b_c, b_c+4

    // fragment gather bases (per documented lane layouts)
    const int fr = lane & 15;
    const int hi = lane >> 4;                    // 0 or 1
    int aB[2], bB[2];
#pragma unroll
    for (int mi = 0; mi < 2; ++mi) aB[mi] = (wm * 32 + mi * 16 + fr) * 72 + hi * 8;
#pragma unroll
    for (int ni = 0; ni < 2; ++ni) bB[ni] = (wn * 32 + ni * 16 + fr) * 72 + hi * 8;

    auto stage = [&](int buf, int kt) {
#pragma unroll
        for (int h = 0; h < 2; ++h) {
            const int ar = a_row + h * 32;
            async_load_b128(lds_addr(&Ash[buf][ar * 72 + a_c * 8]),
                            Ah + (size_t)(m0 + ar) * K + kt + a_c * 8);
        }
#pragma unroll
        for (int h = 0; h < 2; ++h)
#pragma unroll
            for (int c = 0; c < 2; ++c) {
                const int br = b_row + h * 64;
                const int bc = b_c + c * 4;
                async_load_b128(lds_addr(&Bsh[buf][br * 72 + bc * 8]),
                                Bt + (size_t)(n0 + br) * K + kt + bc * 8);
            }
    };

    v8f acc[2][2] = {};
    auto compute = [&](int cur) {
#pragma unroll
        for (int s = 0; s < 2; ++s) {            // two 32-K sub-steps per tile
            const int ks = s * 32;
            v16h af[2], bf[2];
#pragma unroll
            for (int mi = 0; mi < 2; ++mi)
                af[mi] = cat8(*(const v8h*)&Ash[cur][aB[mi] + ks],
                              *(const v8h*)&Ash[cur][aB[mi] + ks + 16]);
#pragma unroll
            for (int ni = 0; ni < 2; ++ni)
                bf[ni] = cat8(*(const v8h*)&Bsh[cur][bB[ni] + ks],
                              *(const v8h*)&Bsh[cur][bB[ni] + ks + 16]);
#pragma unroll
            for (int mi = 0; mi < 2; ++mi)
#pragma unroll
                for (int ni = 0; ni < 2; ++ni)
                    acc[mi][ni] = __builtin_amdgcn_wmma_f32_16x16x32_f16(
                        false, af[mi], false, bf[ni], (short)0,
                        acc[mi][ni], false, false);
        }
    };

    const int niter = K >> 6;                    // K / 64
    stage(0, 0);
    for (int it = 0; it < niter - 1; ++it) {     // steady state: prefetch+compute
        stage((it + 1) & 1, (it + 1) * 64);
        // 6 newest ops in flight for next tile; oldest 6 (cur) must be done
        asm volatile("s_wait_asynccnt 0x6" ::: "memory");
        __syncthreads();                         // whole buffer valid for all waves
        compute(it & 1);
        __syncthreads();                         // reads done before buffer reuse
    }
    asm volatile("s_wait_asynccnt 0x0" ::: "memory");
    __syncthreads();
    compute((niter - 1) & 1);

    // epilogue: documented C/D layout, add bias, store
#pragma unroll
    for (int ni = 0; ni < 2; ++ni) {
        const int   ncol = n0 + wn * 32 + ni * 16 + fr;
        const float bv   = bias[ncol];
#pragma unroll
        for (int mi = 0; mi < 2; ++mi) {
            const int mbase = m0 + wm * 32 + mi * 16 + hi * 8;
#pragma unroll
            for (int i = 0; i < 8; ++i) {
                const float  v   = acc[mi][ni][i] + bv;
                const size_t off = (size_t)(mbase + i) * N + ncol;
                if (OUT_HALF) ((_Float16*)Cout)[off] = (_Float16)v;
                else          ((float*)Cout)[off]    = v;
            }
        }
    }
}

// ---------------------------------------------------------------------------
// launch
// ---------------------------------------------------------------------------
extern "C" void kernel_launch(void* const* d_in, const int* in_sizes, int n_in,
                              void* d_out, int out_size, void* d_ws, size_t ws_size,
                              hipStream_t stream) {
    (void)in_sizes; (void)n_in; (void)out_size; (void)ws_size;

    const float* inputs  = (const float*)d_in[0];
    const float* in_om   = (const float*)d_in[1];
    const float* in_ga   = (const float*)d_in[2];
    const float* in_al   = (const float*)d_in[3];
    const float* proj_W  = (const float*)d_in[4];   // [D][R]
    const float* proj_b  = (const float*)d_in[5];
    const float* res_om  = (const float*)d_in[6];
    const float* res_ga  = (const float*)d_in[7];
    const float* res_al  = (const float*)d_in[8];
    const float* out_om  = (const float*)d_in[9];
    const float* out_ga  = (const float*)d_in[10];
    const float* out_al  = (const float*)d_in[11];
    const float* read_W  = (const float*)d_in[12];  // [R][D]
    const float* read_b  = (const float*)d_in[13];

    float* out    = (float*)d_out;
    float* preds  = out;                                   // [B,T,D]
    float* x_in   = out + (size_t)B_ * T_ * D_;            // [B,T,D]
    float* x_res  = out + 2ull * B_ * T_ * D_;             // [B,T,R]
    float* x_out  = x_res + (size_t)B_ * T_ * R_;          // [B,T,R]

    char* ws = (char*)d_ws;
    const size_t PROJH_BYTES = (size_t)M_ * R_ * 2;        // 128 MB
    const size_t XINH_BYTES  = (size_t)M_ * D_ * 2;        //   8 MB
    _Float16* projWt = (_Float16*)ws;                              // 512 KB
    _Float16* readWt = (_Float16*)(ws + (1u << 20));               // 512 KB
    _Float16* proj_h = (_Float16*)(ws + (2u << 20));               // 128 MB
    _Float16* x_in_h = (_Float16*)(ws + (2u << 20) + PROJH_BYTES); //   8 MB
    _Float16* x_out_h= (_Float16*)(ws + (2u << 20) + PROJH_BYTES + XINH_BYTES); // 128 MB

    // K0: one-time weight convert+transpose to [N][K] f16
    convert_transpose_f16<<<(D_ * R_ + 255) / 256, 256, 0, stream>>>(proj_W, projWt, D_, R_);
    convert_transpose_f16<<<(R_ * D_ + 255) / 256, 256, 0, stream>>>(read_W, readWt, R_, D_);

    // K1: input oscillator scan -> x_in_all (fp32) + x_in_h (f16)
    scan_in_kernel<<<B_, 128, 0, stream>>>(inputs, in_om, in_ga, in_al, x_in, x_in_h);

    // K2: proj = x_in @ proj_W + proj_b   ([32768,128]@[128,2048], f16 out)
    wmma_gemm_kernel<true><<<dim3(R_ / 128, M_ / 64), 256, 0, stream>>>(
        x_in_h, projWt, proj_b, (void*)proj_h, D_, R_);

    // K3: fused reservoir + output oscillator scans
    scan_res_out_kernel<<<dim3(R_ / 256, B_), 256, 0, stream>>>(
        proj_h, res_om, res_ga, res_al, out_om, out_ga, out_al,
        x_res, x_out, x_out_h);

    // K4: preds = x_out @ read_W + read_b ([32768,2048]@[2048,128], f32 out)
    wmma_gemm_kernel<false><<<dim3(D_ / 128, M_ / 64), 256, 0, stream>>>(
        x_out_h, readWt, read_b, (void*)preds, R_, D_);
}
